// SetAbstraction_7834020348220
// MI455X (gfx1250) — matrix operand order repaired and checked by reading, not compiled, not run on hardware
//
#include <hip/hip_runtime.h>
#include <hip/hip_bf16.h>

#define BB 8
#define NN 8192
#define SS 2048
#define KK 32

typedef _Float16 half_t;
typedef half_t v16h __attribute__((ext_vector_type(16)));
typedef half_t v8h  __attribute__((ext_vector_type(8)));
typedef float  v8f  __attribute__((ext_vector_type(8)));

// ---------------------------------------------------------------------------
// Kernel 1: Farthest Point Sampling.  One block per batch, 1024 threads,
// 8 points per thread held in registers (coords + running min-distance).
// Per iteration: VALU distance update, wave32 shuffle argmax (value,idx,xyz),
// then 32 wave results reduced by wave 0 through LDS. Tie-break = lower index
// to match jnp.argmax. Writes new_xyz directly in [B,3,S] output layout.
// ---------------------------------------------------------------------------
__global__ __launch_bounds__(1024) void fps_kernel(const float* __restrict__ xyz,
                                                   float* __restrict__ out_newxyz) {
  const int b   = blockIdx.x;
  const int tid = threadIdx.x;
  const float* xs = xyz + (size_t)b * 3 * NN;
  const float* ys = xs + NN;
  const float* zs = ys + NN;

  float px[8], py[8], pz[8], pd[8];
#pragma unroll
  for (int k = 0; k < 8; k++) {
    int i = k * 1024 + tid;             // coalesced
    px[k] = xs[i]; py[k] = ys[i]; pz[k] = zs[i];
    pd[k] = 1e10f;
  }

  __shared__ float rV[32]; __shared__ int rI[32];
  __shared__ float rX[32]; __shared__ float rY[32]; __shared__ float rZ[32];
  __shared__ float cX, cY, cZ;

  if (tid == 0) { cX = px[0]; cY = py[0]; cZ = pz[0]; }   // first centroid = point 0
  __syncthreads();

  const int lane = tid & 31;
  const int wave = tid >> 5;
  const size_t obase = (size_t)b * 3 * SS;

  for (int s = 0; s < SS; s++) {
    const float ccx = cX, ccy = cY, ccz = cZ;
    if (tid == 0) {
      out_newxyz[obase + 0 * SS + s] = ccx;
      out_newxyz[obase + 1 * SS + s] = ccy;
      out_newxyz[obase + 2 * SS + s] = ccz;
    }

    float bv = -1.0f, bx = 0.f, by = 0.f, bz = 0.f; int bi = 0x7fffffff;
#pragma unroll
    for (int k = 0; k < 8; k++) {
      float dx = px[k] - ccx, dy = py[k] - ccy, dz = pz[k] - ccz;
      float d = dx * dx + dy * dy + dz * dz;
      pd[k] = fminf(pd[k], d);
      int i = k * 1024 + tid;
      bool take = (pd[k] > bv) || (pd[k] == bv && i < bi);
      if (take) { bv = pd[k]; bi = i; bx = px[k]; by = py[k]; bz = pz[k]; }
    }
    // wave32 reduction carrying (val, idx, x, y, z)
#pragma unroll
    for (int off = 16; off >= 1; off >>= 1) {
      float ov = __shfl_down(bv, off, 32);
      int   oi = __shfl_down(bi, off, 32);
      float ox = __shfl_down(bx, off, 32);
      float oy = __shfl_down(by, off, 32);
      float oz = __shfl_down(bz, off, 32);
      bool take = (ov > bv) || (ov == bv && oi < bi);
      if (take) { bv = ov; bi = oi; bx = ox; by = oy; bz = oz; }
    }
    __syncthreads();
    if (lane == 0) { rV[wave] = bv; rI[wave] = bi; rX[wave] = bx; rY[wave] = by; rZ[wave] = bz; }
    __syncthreads();
    if (wave == 0) {
      bv = rV[lane]; bi = rI[lane]; bx = rX[lane]; by = rY[lane]; bz = rZ[lane];
#pragma unroll
      for (int off = 16; off >= 1; off >>= 1) {
        float ov = __shfl_down(bv, off, 32);
        int   oi = __shfl_down(bi, off, 32);
        float ox = __shfl_down(bx, off, 32);
        float oy = __shfl_down(by, off, 32);
        float oz = __shfl_down(bz, off, 32);
        bool take = (ov > bv) || (ov == bv && oi < bi);
        if (take) { bv = ov; bi = oi; bx = ox; by = oy; bz = oz; }
      }
      if (lane == 0) { cX = bx; cY = by; cZ = bz; }
    }
    __syncthreads();
  }
}

// ---------------------------------------------------------------------------
// Kernel 2: kNN (top-K=32 smallest squared distances among N points).
// One thread per centroid; points streamed through a 2048-point LDS tile
// (all lanes read the same LDS word -> broadcast, no bank conflicts).
// Register-resident unordered top-K: replace-current-max only when beaten
// (~K*ln(N/K) ~ 180 replacements out of 8192), fully unrolled so the
// 32-entry lists stay in VGPRs. Order within K is irrelevant (max-pool later).
// ---------------------------------------------------------------------------
#define TILE 2048
__global__ __launch_bounds__(256) void knn_kernel(const float* __restrict__ xyz,
                                                  const float* __restrict__ newxyz,
                                                  int* __restrict__ idx_out) {
  const int gid = blockIdx.x * 256 + threadIdx.x;  // centroid id, same b per block
  const int b = gid / SS;
  const int s = gid - b * SS;
  const size_t nb = (size_t)b * 3 * SS;
  const float cx = newxyz[nb + s];
  const float cy = newxyz[nb + SS + s];
  const float cz = newxyz[nb + 2 * SS + s];

  __shared__ float sx[TILE], sy[TILE], sz[TILE];
  const float* xs = xyz + (size_t)b * 3 * NN;
  const float* ys = xs + NN;
  const float* zs = ys + NN;

  float bd[KK]; int bi[KK];
#pragma unroll
  for (int t = 0; t < KK; t++) { bd[t] = 3.0e38f; bi[t] = 0; }
  float wmax = 3.0e38f;

  for (int t0 = 0; t0 < NN; t0 += TILE) {
    __syncthreads();
    for (int j = threadIdx.x; j < TILE; j += 256) {
      sx[j] = xs[t0 + j]; sy[j] = ys[t0 + j]; sz[j] = zs[t0 + j];
    }
    __syncthreads();
#pragma unroll 4
    for (int j = 0; j < TILE; j++) {
      float dx = sx[j] - cx, dy = sy[j] - cy, dz = sz[j] - cz;
      float d = dx * dx + dy * dy + dz * dz;
      if (d < wmax) {
        int jj = t0 + j;
        bool done = false;
#pragma unroll
        for (int t = 0; t < KK; t++) {
          if (!done && bd[t] == wmax) { bd[t] = d; bi[t] = jj; done = true; }
        }
        wmax = bd[0];
#pragma unroll
        for (int t = 1; t < KK; t++) wmax = fmaxf(wmax, bd[t]);
      }
    }
  }
  const int base = gid * KK;
#pragma unroll
  for (int t = 0; t < KK; t++) idx_out[base + t] = bi[t];
}

// ---------------------------------------------------------------------------
// Kernel 3: grouped MLP + maxpool with v_wmma_f32_16x16x32_f16.
// 128 threads = 4 waves, one centroid per wave. Layer1 (K=3) in VALU -> h1
// (32x64 f16) in per-wave LDS; layers 2/3 as WMMA GEMMs with fragments loaded
// in contiguous k order (k = 16*(lane/16) + s) for A (16x32) / B (32x16), f32 C per its layout.
// Weights staged once per block as f16 row-major [O][K] so a B-fragment is one
// contiguous 32B LDS read; an A-fragment is two 16B reads. Max-pool over the
// 32 rows via per-lane max + __shfl_xor(16) across the lane halves.
// ---------------------------------------------------------------------------
__device__ __forceinline__ v16h loadA(const half_t* Hm, int mt, int kc, int lm, int lh) {
  const half_t* p = Hm + (mt * 16 + lm) * 64 + kc * 32 + lh * 16;
  v8h lo = *(const v8h*)p;          // K = kbase + 0..7
  v8h hi = *(const v8h*)(p + 8);    // K = kbase + 8..15
  return __builtin_shufflevector(lo, hi, 0,1,2,3,4,5,6,7,8,9,10,11,12,13,14,15);
}

__global__ __launch_bounds__(128) void mlp_kernel(const float* __restrict__ xyz,
                                                  const float* __restrict__ newxyz,
                                                  const int* __restrict__ knn_idx,
                                                  const float* __restrict__ W0, const float* __restrict__ b0,
                                                  const float* __restrict__ W1, const float* __restrict__ b1,
                                                  const float* __restrict__ W2, const float* __restrict__ b2,
                                                  float* __restrict__ out_pts) {
  __shared__ float sW0[64 * 3];
  __shared__ float sb0[64], sb1[64], sb2[128];
  __shared__ alignas(32) half_t sW1[64 * 64];    // [o][k]
  __shared__ alignas(32) half_t sW2[128 * 64];   // [o][k]
  __shared__ alignas(32) half_t h1[4][32 * 64];  // per-wave activations
  __shared__ alignas(32) half_t h2[4][32 * 64];

  const int tid = threadIdx.x;
  for (int i = tid; i < 192; i += 128) sW0[i] = W0[i];
  if (tid < 64)  sb0[tid] = b0[tid];
  if (tid < 64)  sb1[tid] = b1[tid];
  if (tid < 128) sb2[tid] = b2[tid];
  for (int i = tid; i < 64 * 64; i += 128)  sW1[i] = (half_t)W1[i];
  for (int i = tid; i < 128 * 64; i += 128) sW2[i] = (half_t)W2[i];
  __syncthreads();

  const int wave = tid >> 5, lane = tid & 31;
  const int gid = blockIdx.x * 4 + wave;      // centroid id in [0, B*S)
  const int b = gid / SS;
  const int s = gid - b * SS;

  const size_t nb = (size_t)b * 3 * SS;
  const float cx = newxyz[nb + s];
  const float cy = newxyz[nb + SS + s];
  const float cz = newxyz[nb + 2 * SS + s];

  // gather: lane p = point p of the group
  const int pid = knn_idx[gid * KK + lane];
  const float* xs = xyz + (size_t)b * 3 * NN;
  const float gx = xs[pid] - cx;
  const float gy = xs[NN + pid] - cy;
  const float gz = xs[2 * NN + pid] - cz;

  // ---- layer 1: 3 -> 64 (VALU), h1 row p = lane p ----
  half_t* H1 = h1[wave];
#pragma unroll 4
  for (int o = 0; o < 64; o++) {
    float v = fmaf(gx, sW0[o * 3 + 0],
              fmaf(gy, sW0[o * 3 + 1],
              fmaf(gz, sW0[o * 3 + 2], sb0[o])));
    H1[lane * 64 + o] = (half_t)fmaxf(v, 0.0f);
  }
  __syncthreads();

  const int lm = lane & 15;   // column / row-within-tile
  const int lh = lane >> 4;   // lane-half select

  // ---- layer 2: [32x64] x [64x64] via WMMA ----
  half_t* H2 = h2[wave];
#pragma unroll
  for (int n = 0; n < 4; n++) {
    v8f c0 = {}; v8f c1 = {};
#pragma unroll
    for (int kc = 0; kc < 2; kc++) {
      v16h bf = *(const v16h*)&sW1[(n * 16 + lm) * 64 + kc * 32 + lh * 16];
      v16h a0 = loadA(H1, 0, kc, lm, lh);
      v16h a1 = loadA(H1, 1, kc, lm, lh);
      c0 = __builtin_amdgcn_wmma_f32_16x16x32_f16(false, a0, false, bf, (short)0, c0, false, false);
      c1 = __builtin_amdgcn_wmma_f32_16x16x32_f16(false, a1, false, bf, (short)0, c1, false, false);
    }
    const float bias = sb1[n * 16 + lm];
#pragma unroll
    for (int r = 0; r < 8; r++) {
      int row0 = lh * 8 + r;          // m-tile 0
      int row1 = 16 + lh * 8 + r;     // m-tile 1
      H2[row0 * 64 + n * 16 + lm] = (half_t)fmaxf(c0[r] + bias, 0.0f);
      H2[row1 * 64 + n * 16 + lm] = (half_t)fmaxf(c1[r] + bias, 0.0f);
    }
  }
  __syncthreads();

  // ---- layer 3: [32x64] x [64x128] via WMMA, fused max-pool over 32 rows ----
  const size_t obase = (size_t)b * 128 * SS + s;
#pragma unroll
  for (int n = 0; n < 8; n++) {
    v8f c0 = {}; v8f c1 = {};
#pragma unroll
    for (int kc = 0; kc < 2; kc++) {
      v16h bf = *(const v16h*)&sW2[(n * 16 + lm) * 64 + kc * 32 + lh * 16];
      v16h a0 = loadA(H2, 0, kc, lm, lh);
      v16h a1 = loadA(H2, 1, kc, lm, lh);
      c0 = __builtin_amdgcn_wmma_f32_16x16x32_f16(false, a0, false, bf, (short)0, c0, false, false);
      c1 = __builtin_amdgcn_wmma_f32_16x16x32_f16(false, a1, false, bf, (short)0, c1, false, false);
    }
    const float bias = sb2[n * 16 + lm];
    float m = -3.0e38f;
#pragma unroll
    for (int r = 0; r < 8; r++) m = fmaxf(m, fmaxf(c0[r], c1[r]));
    m = fmaxf(m + bias, 0.0f);                     // relu(max) == max(relu)
    m = fmaxf(m, __shfl_xor(m, 16, 32));           // combine lane halves -> all 32 rows
    if (lh == 0) out_pts[obase + (size_t)(n * 16 + lm) * SS] = m;
  }
}

// ---------------------------------------------------------------------------
extern "C" void kernel_launch(void* const* d_in, const int* in_sizes, int n_in,
                              void* d_out, int out_size, void* d_ws, size_t ws_size,
                              hipStream_t stream) {
  const float* xyz = (const float*)d_in[0];
  const float* W0  = (const float*)d_in[1];
  const float* b0  = (const float*)d_in[2];
  const float* W1  = (const float*)d_in[3];
  const float* b1  = (const float*)d_in[4];
  const float* W2  = (const float*)d_in[5];
  const float* b2  = (const float*)d_in[6];

  float* out      = (float*)d_out;
  float* newxyz   = out;                      // [B,3,S]  (49152 floats)
  float* newpts   = out + (size_t)BB * 3 * SS; // [B,128,S]
  int*   idx_ws   = (int*)d_ws;               // [B*S, K] int32 (2 MB)

  fps_kernel<<<dim3(BB), dim3(1024), 0, stream>>>(xyz, newxyz);
  knn_kernel<<<dim3((BB * SS) / 256), dim3(256), 0, stream>>>(xyz, newxyz, idx_ws);
  mlp_kernel<<<dim3((BB * SS) / 4), dim3(128), 0, stream>>>(xyz, newxyz, idx_ws,
                                                            W0, b0, W1, b1, W2, b2, newpts);
}
